// TimeSeriesRNNRepresentation_74062416053139
// MI455X (gfx1250) — compile-verified, matrix operand order checked by reading, not run on hardware
//
#include <hip/hip_runtime.h>
#include <hip/hip_bf16.h>
#include <math.h>

typedef __attribute__((ext_vector_type(16))) _Float16 v16h;
typedef __attribute__((ext_vector_type(8)))  _Float16 v8h;
typedef __attribute__((ext_vector_type(8)))  float    v8f;
typedef __attribute__((ext_vector_type(4)))  float    v4f;
typedef __attribute__((ext_vector_type(2)))  float    v2f;

#define B_  64
#define T_  512
#define D_  2048
#define H1_ 128
#define H2_ 64

#define WMMA_F32(a, b, c) \
    __builtin_amdgcn_wmma_f32_16x16x4_f32(false, (a), false, (b), (short)0, (c), false, false)

// ---------------------------------------------------------------------------
// Kernel 1: convert w_ih1 (fp32, [H1, D]) -> f16 once (reused by 2048 blocks)
// ---------------------------------------------------------------------------
__global__ void cvt_w_kernel(const float* __restrict__ w, _Float16* __restrict__ w16, int n) {
    int i = blockIdx.x * blockDim.x + threadIdx.x;
    if (i < n) w16[i] = (_Float16)w[i];
}

// ---------------------------------------------------------------------------
// Kernel 2: z1 = x @ w_ih1^T + (b_ih1 + b_hh1)   [B*T, H1]
// Block = 128 threads = 4 waves; block computes 16 rows x 128 cols.
// Wave w owns TWO N-tiles (n0 = w*32, w*32+16): halves redundant A traffic and
// gives two independent WMMA accumulation chains.
// A layout per ISA (16-bit A 16x32): lane holds row (lane&15),
//   K segments [koff, koff+8) and [koff+16, koff+24), koff = (lane>>4)*8.
// ---------------------------------------------------------------------------
__global__ __launch_bounds__(128) void gemm1_kernel(const float* __restrict__ x,
                                                    const _Float16* __restrict__ w16,
                                                    const float* __restrict__ b_ih1,
                                                    const float* __restrict__ b_hh1,
                                                    float* __restrict__ z1) {
    const int lane = threadIdx.x & 31;
    const int wv   = threadIdx.x >> 5;       // 0..3
    const int m0   = blockIdx.x * 16;
    const int nA0  = wv * 32;
    const int nB0  = nA0 + 16;
    const int hf   = lane >> 4;              // half-wave
    const int l15  = lane & 15;
    const int koff = hf * 8;

    const float*    xp  = x   + (size_t)(m0 + l15) * D_;
    const _Float16* wpA = w16 + (size_t)(nA0 + l15) * D_;
    const _Float16* wpB = w16 + (size_t)(nB0 + l15) * D_;

    v8f cA = {}, cB = {};
    for (int k0 = 0; k0 < D_; k0 += 32) {
        const float* xa = xp + k0 + koff;
        __builtin_prefetch(xa + 1024, 0, 0);              // stream-ahead (global_prefetch)
        v4f f0 = *(const v4f*)(xa);
        v4f f1 = *(const v4f*)(xa + 4);
        v4f f2 = *(const v4f*)(xa + 16);
        v4f f3 = *(const v4f*)(xa + 20);
        v16h a;
#pragma unroll
        for (int i = 0; i < 4; ++i) {
            a[i]      = (_Float16)f0[i];
            a[4 + i]  = (_Float16)f1[i];
            a[8 + i]  = (_Float16)f2[i];
            a[12 + i] = (_Float16)f3[i];
        }
        v8h bA0 = *(const v8h*)(wpA + k0 + koff);
        v8h bA1 = *(const v8h*)(wpA + k0 + koff + 16);
        v8h bB0 = *(const v8h*)(wpB + k0 + koff);
        v8h bB1 = *(const v8h*)(wpB + k0 + koff + 16);
        v16h bA, bB;
#pragma unroll
        for (int i = 0; i < 8; ++i) {
            bA[i] = bA0[i]; bA[8 + i] = bA1[i];
            bB[i] = bB0[i]; bB[8 + i] = bB1[i];
        }
        cA = __builtin_amdgcn_wmma_f32_16x16x32_f16(false, a, false, bA, (short)0, cA, false, false);
        cB = __builtin_amdgcn_wmma_f32_16x16x32_f16(false, a, false, bB, (short)0, cB, false, false);
    }

    const int colA = nA0 + l15, colB = nB0 + l15;
    const float biasA = b_ih1[colA] + b_hh1[colA];
    const float biasB = b_ih1[colB] + b_hh1[colB];
#pragma unroll
    for (int r = 0; r < 8; ++r) {
        const int row = m0 + r + hf * 8;                  // C/D layout: M = r + (lane>=16)*8
        z1[(size_t)row * H1_ + colA] = cA[r] + biasA;
        z1[(size_t)row * H1_ + colB] = cB[r] + biasB;
    }
}

// ---------------------------------------------------------------------------
// Kernel 3: recurrent scan. 4 blocks x 16 batch rows, T=512 steps.
// Software-pipelined: waves 0-3 do layer 1 for step t (two 16-col tiles each),
// waves 4-7 do layer 2 for step t-1 concurrently. Double-buffered h1/h2 in
// LDS -> ONE barrier per step. K split into independent WMMA chains (depth 16).
// ---------------------------------------------------------------------------
#define S1 132   // padded K-stride, K=128
#define S2 68    // padded K-stride, K=64

__global__ __launch_bounds__(256) void scan_kernel(const float* __restrict__ z1,
                                                   const float* __restrict__ w_hh1,
                                                   const float* __restrict__ w_ih2,
                                                   const float* __restrict__ w_hh2,
                                                   const float* __restrict__ b_ih2,
                                                   const float* __restrict__ b_hh2,
                                                   float* __restrict__ out) {
    __shared__ float sW1[H1_ * S1];       // w_hh1[n][k]
    __shared__ float sW2[H2_ * S1];       // w_ih2[n][k]
    __shared__ float sW3[H2_ * S2];       // w_hh2[n][k]
    __shared__ float sH1[2 * 16 * S1];    // double-buffered h1 [m][k]
    __shared__ float sH2[2 * 16 * S2];    // double-buffered h2 [m][k]

    const int tid = threadIdx.x;
    for (int i = tid; i < H1_ * H1_; i += 256) sW1[(i >> 7) * S1 + (i & 127)] = w_hh1[i];
    for (int i = tid; i < H2_ * H1_; i += 256) sW2[(i >> 7) * S1 + (i & 127)] = w_ih2[i];
    for (int i = tid; i < H2_ * H2_; i += 256) sW3[(i >> 6) * S2 + (i & 63)]  = w_hh2[i];
    for (int i = tid; i < 2 * 16 * S1; i += 256) sH1[i] = 0.f;
    for (int i = tid; i < 2 * 16 * S2; i += 256) sH2[i] = 0.f;
    __syncthreads();

    const int lane = tid & 31;
    const int wv   = tid >> 5;
    const int hf   = lane >> 4;
    const int l15  = lane & 15;
    const int b0   = blockIdx.x * 16;
    const int kq   = hf * 2;              // f32 A/B layout: K = k + (lane>>4)*2 + {0,1}

    // group A (waves 0-3): layer-1 columns [wv*32, wv*32+32)
    const int nA = (wv & 3) * 32 + l15;
    const int nB = nA + 16;
    // group B (waves 4-7): layer-2 columns [(wv-4)*16, ...)
    const int n2 = (wv & 3) * 16 + l15;
    const float bias2 = b_ih2[n2] + b_hh2[n2];

    // z1 addressing: element ((b0+m)*T + t)*H1 + col,  m = r + hf*8
    const float* zp = z1 + (size_t)(b0 + hf * 8) * T_ * H1_;

    float zA[8], zB[8];
    if (wv < 4) {
#pragma unroll
        for (int r = 0; r < 8; ++r) {                    // z(0)
            zA[r] = zp[((size_t)r * T_) * H1_ + nA];
            zB[r] = zp[((size_t)r * T_) * H1_ + nB];
        }
    }

    for (int t = 0; t < T_; ++t) {
        if (wv < 4) {
            // prefetch z(t+1) (clamped; hides under the WMMA chain)
            const int tn = (t + 1 < T_) ? t + 1 : t;
            float zAn[8], zBn[8];
#pragma unroll
            for (int r = 0; r < 8; ++r) {
                zAn[r] = zp[((size_t)r * T_ + tn) * H1_ + nA];
                zBn[r] = zp[((size_t)r * T_ + tn) * H1_ + nB];
            }
            // layer 1: h1(t) = tanh(z(t) + h1(t-1) @ w_hh1^T)
            const float* h1p = sH1 + (t & 1) * (16 * S1);
            float*       h1w = sH1 + ((t + 1) & 1) * (16 * S1);
            v8f cAlo = {}, cAhi = {}, cBlo = {}, cBhi = {};
#pragma unroll
            for (int k = 0; k < 64; k += 4) {
                v2f alo = *(const v2f*)(h1p + l15 * S1 + k + kq);
                v2f ahi = *(const v2f*)(h1p + l15 * S1 + 64 + k + kq);
                v2f bAl = *(const v2f*)(sW1 + nA * S1 + k + kq);
                v2f bAh = *(const v2f*)(sW1 + nA * S1 + 64 + k + kq);
                v2f bBl = *(const v2f*)(sW1 + nB * S1 + k + kq);
                v2f bBh = *(const v2f*)(sW1 + nB * S1 + 64 + k + kq);
                cAlo = WMMA_F32(alo, bAl, cAlo);
                cAhi = WMMA_F32(ahi, bAh, cAhi);
                cBlo = WMMA_F32(alo, bBl, cBlo);
                cBhi = WMMA_F32(ahi, bBh, cBhi);
            }
#pragma unroll
            for (int r = 0; r < 8; ++r) {
                h1w[(r + hf * 8) * S1 + nA] = tanhf(cAlo[r] + cAhi[r] + zA[r]);
                h1w[(r + hf * 8) * S1 + nB] = tanhf(cBlo[r] + cBhi[r] + zB[r]);
                zA[r] = zAn[r];
                zB[r] = zBn[r];
            }
        } else if (t > 0) {
            // layer 2 for step t-1: h2(t-1) = tanh(h1(t-1)@w_ih2^T + h2(t-2)@w_hh2^T + b2)
            const float* h1p = sH1 + (t & 1) * (16 * S1);        // h1(t-1)
            const float* h2p = sH2 + (t & 1) * (16 * S2);        // h2(t-2)
            float*       h2w = sH2 + ((t + 1) & 1) * (16 * S2);  // h2(t-1)
            v8f c0 = {}, c1 = {}, c2 = {};
#pragma unroll
            for (int k = 0; k < 64; k += 4) {
                v2f a0 = *(const v2f*)(h1p + l15 * S1 + k + kq);
                v2f a1 = *(const v2f*)(h1p + l15 * S1 + 64 + k + kq);
                v2f a2 = *(const v2f*)(h2p + l15 * S2 + k + kq);
                v2f b0v = *(const v2f*)(sW2 + n2 * S1 + k + kq);
                v2f b1v = *(const v2f*)(sW2 + n2 * S1 + 64 + k + kq);
                v2f b2v = *(const v2f*)(sW3 + n2 * S2 + k + kq);
                c0 = WMMA_F32(a0, b0v, c0);
                c1 = WMMA_F32(a1, b1v, c1);
                c2 = WMMA_F32(a2, b2v, c2);
            }
#pragma unroll
            for (int r = 0; r < 8; ++r)
                h2w[(r + hf * 8) * S2 + n2] = tanhf(c0[r] + c1[r] + c2[r] + bias2);
        }
        __syncthreads();   // single barrier per step (orders all buffer hand-offs)
    }

    // epilogue: waves 4-7 compute h2(T-1) from h1(T-1)=sH1[0], h2(T-2)=sH2[0]
    if (wv >= 4) {
        const float* h1p = sH1;      // T_ even -> buffer 0
        const float* h2p = sH2;
        v8f c0 = {}, c1 = {}, c2 = {};
#pragma unroll
        for (int k = 0; k < 64; k += 4) {
            v2f a0 = *(const v2f*)(h1p + l15 * S1 + k + kq);
            v2f a1 = *(const v2f*)(h1p + l15 * S1 + 64 + k + kq);
            v2f a2 = *(const v2f*)(h2p + l15 * S2 + k + kq);
            v2f b0v = *(const v2f*)(sW2 + n2 * S1 + k + kq);
            v2f b1v = *(const v2f*)(sW2 + n2 * S1 + 64 + k + kq);
            v2f b2v = *(const v2f*)(sW3 + n2 * S2 + k + kq);
            c0 = WMMA_F32(a0, b0v, c0);
            c1 = WMMA_F32(a1, b1v, c1);
            c2 = WMMA_F32(a2, b2v, c2);
        }
#pragma unroll
        for (int r = 0; r < 8; ++r)
            out[(size_t)(b0 + r + hf * 8) * H2_ + n2] = tanhf(c0[r] + c1[r] + c2[r] + bias2);
    }
}

// ---------------------------------------------------------------------------
extern "C" void kernel_launch(void* const* d_in, const int* in_sizes, int n_in,
                              void* d_out, int out_size, void* d_ws, size_t ws_size,
                              hipStream_t stream) {
    const float* x     = (const float*)d_in[0];
    const float* w_ih1 = (const float*)d_in[1];
    const float* w_hh1 = (const float*)d_in[2];
    const float* b_ih1 = (const float*)d_in[3];
    const float* b_hh1 = (const float*)d_in[4];
    const float* w_ih2 = (const float*)d_in[5];
    const float* w_hh2 = (const float*)d_in[6];
    const float* b_ih2 = (const float*)d_in[7];
    const float* b_hh2 = (const float*)d_in[8];

    _Float16* w16 = (_Float16*)d_ws;                                             // 512 KB
    float*    z1  = (float*)((char*)d_ws + (size_t)H1_ * D_ * sizeof(_Float16)); // 16 MB

    cvt_w_kernel<<<(H1_ * D_ + 255) / 256, 256, 0, stream>>>(w_ih1, w16, H1_ * D_);
    gemm1_kernel<<<(B_ * T_) / 16, 128, 0, stream>>>(x, w16, b_ih1, b_hh1, z1);
    scan_kernel<<<B_ / 16, 256, 0, stream>>>(z1, w_hh1, w_ih2, w_hh2,
                                             b_ih2, b_hh2, (float*)d_out);
}
